// DSSAttention_65661460021374
// MI455X (gfx1250) — compile-verified, matrix operand order checked by reading, not run on hardware
//
#include <hip/hip_runtime.h>
#include <hip/hip_bf16.h>
#include <cstdint>
#include <cstddef>

#define NTOK  4096
#define CDIM  512
#define NHEAD 8
#define HDIM  64
#define RANK  32

typedef __attribute__((ext_vector_type(16))) _Float16 v16h;
typedef __attribute__((ext_vector_type(8)))  _Float16 v8h;
typedef __attribute__((ext_vector_type(8)))  float    v8f;

// ---------------------------------------------------------------------------
// WMMA helpers (gfx1250, wave32). A-matrix 16x32 f16 fragment per ISA layout:
//   lanes 0-15  (half=0): row M = lane,   halves K = 0..7 and 16..23
//   lanes 16-31 (half=1): row M = lane-16, halves K = 8..15 and 24..31
// B fragment for C = A * B^T is loaded identically from the [N,K] row-major
// matrix (column-major B == row-major B^T).
// ---------------------------------------------------------------------------
static __device__ __forceinline__ v16h load_frag_row(const _Float16* __restrict__ p,
                                                     int halfSel) {
  v8h a = *(const v8h*)(p + halfSel * 8);
  v8h b = *(const v8h*)(p + 16 + halfSel * 8);
  v16h r;
#pragma unroll
  for (int i = 0; i < 8; ++i) { r[i] = a[i]; r[i + 8] = b[i]; }
  return r;
}

static __device__ __forceinline__ v8f wmma_f16(v16h a, v16h b, v8f c) {
  // (neg_a, A, neg_b, B, c_mod, C, reuse_a, reuse_b)
  return __builtin_amdgcn_wmma_f32_16x16x32_f16(false, a, false, b, (short)0, c,
                                                false, false);
}

// ---------------------------------------------------------------------------
// Tiled GEMM: C[M,N] = A[M,K] (f16, row-major) * Bt[N,K]^T (f16, row-major =
// torch weight layout).  One wave computes a 16 x (16*NT) strip: the A
// fragment is loaded once per 32-wide k-step and reused across NT WMMAs
// ((1+NT) fragment loads per NT WMMAs), with NT independent accumulators so
// the scheduler can hide WMMA latency.  Optional fused bias[N], residual[M,N]
// and f16 shadow copy of C.
// ---------------------------------------------------------------------------
template <int NT>
__global__ void __launch_bounds__(32)
gemm_wmma_nt(const _Float16* __restrict__ A, const _Float16* __restrict__ Bt,
             float* __restrict__ C, _Float16* __restrict__ Ch,
             const float* __restrict__ bias, const float* __restrict__ resid,
             int M, int N, int K)
{
  const int lane = threadIdx.x & 31;
  const int half = lane >> 4;
  const int r    = lane & 15;
  const int row0 = blockIdx.x * 16;
  const int col0 = blockIdx.y * (16 * NT);

  const _Float16* arow = A + (size_t)(row0 + r) * K;
  const _Float16* brow[NT];
#pragma unroll
  for (int t = 0; t < NT; ++t)
    brow[t] = Bt + (size_t)(col0 + t * 16 + r) * K;

  v8f acc[NT];
#pragma unroll
  for (int t = 0; t < NT; ++t) acc[t] = (v8f){};

  for (int k0 = 0; k0 < K; k0 += 32) {
    __builtin_prefetch(arow + k0 + 64, 0, 1);  // global_prefetch_b8
    v16h a = load_frag_row(arow + k0, half);
#pragma unroll
    for (int t = 0; t < NT; ++t) {
      v16h b = load_frag_row(brow[t] + k0, half);
      acc[t] = wmma_f16(a, b, acc[t]);
    }
  }

#pragma unroll
  for (int t = 0; t < NT; ++t) {
#pragma unroll
    for (int i = 0; i < 8; ++i) {
      int row = row0 + half * 8 + i;   // C layout: VGPR i -> M = 8*half + i
      int col = col0 + t * 16 + r;     // N = lane & 15
      float v = acc[t][i];
      if (bias)  v += bias[col];
      if (resid) v += resid[(size_t)row * N + col];
      size_t o = (size_t)row * N + col;
      C[o] = v;
      if (Ch) Ch[o] = (_Float16)v;
    }
  }
}

// ---------------------------------------------------------------------------
// Attention pass 1: full-row softmax statistics (max, sum) with online
// (flash-style) streaming over all key blocks; never materializes the
// 4096x4096 score matrix (512 MB > 192 MB L2). One wave per (head, q block).
// q fragments stay in registers for the whole 4096-key sweep.
// ---------------------------------------------------------------------------
__global__ void __launch_bounds__(32)
attn_stats(const _Float16* __restrict__ qh, const _Float16* __restrict__ kh,
           float* __restrict__ rowmax, float* __restrict__ rowsum)
{
  const int lane = threadIdx.x & 31;
  const int half = lane >> 4;
  const int r    = lane & 15;
  const int head = blockIdx.y;
  const int qb   = blockIdx.x;

  const _Float16* qrow = qh + ((size_t)head * NTOK + qb * 16 + r) * HDIM;
  v16h qa0 = load_frag_row(qrow,      half);   // K = 0..31
  v16h qa1 = load_frag_row(qrow + 32, half);   // K = 32..63

  float m[8], s[8];
#pragma unroll
  for (int i = 0; i < 8; ++i) { m[i] = -1e30f; s[i] = 0.f; }

  const _Float16* khead = kh + (size_t)head * NTOK * HDIM;
  for (int kb = 0; kb < NTOK / 16; ++kb) {
    const _Float16* krow = khead + (size_t)(kb * 16 + r) * HDIM;
    __builtin_prefetch(krow + 16 * HDIM, 0, 1);
    v16h b0 = load_frag_row(krow,      half);
    v16h b1 = load_frag_row(krow + 32, half);
    v8f c = {};
    c = wmma_f16(qa0, b0, c);
    c = wmma_f16(qa1, b1, c);
#pragma unroll
    for (int i = 0; i < 8; ++i) {           // online max/sum per lane slot
      float x  = c[i];
      float mn = fmaxf(m[i], x);
      s[i] = s[i] * __expf(m[i] - mn) + __expf(x - mn);
      m[i] = mn;
    }
  }

  // Merge the 16 lanes of each half (same rows, different key columns).
#pragma unroll
  for (int off = 1; off < 16; off <<= 1) {
#pragma unroll
    for (int i = 0; i < 8; ++i) {
      float mo = __shfl_xor(m[i], off, 32);
      float so = __shfl_xor(s[i], off, 32);
      float mn = fmaxf(m[i], mo);
      s[i] = s[i] * __expf(m[i] - mn) + so * __expf(mo - mn);
      m[i] = mn;
    }
  }
  if (r == 0) {
#pragma unroll
    for (int i = 0; i < 8; ++i) {
      int row = qb * 16 + half * 8 + i;
      rowmax[(size_t)head * NTOK + row] = m[i];
      rowsum[(size_t)head * NTOK + row] = s[i];
    }
  }
}

// ---------------------------------------------------------------------------
// Attention pass 2: only the diagonal 16x16 block survives the block-causal
// mask.  Recompute it, normalize with the full-row stats, apply in-block
// causal mask, and do the tiny P(16x16) * V(16x64) through LDS.
// ---------------------------------------------------------------------------
__global__ void __launch_bounds__(32)
attn_diag_wv(const _Float16* __restrict__ qh, const _Float16* __restrict__ kh,
             const float* __restrict__ vh, const float* __restrict__ rowmax,
             const float* __restrict__ rowsum, float* __restrict__ wv)
{
  __shared__ float P[16][16];
  __shared__ float V[16][HDIM];

  const int lane = threadIdx.x & 31;
  const int half = lane >> 4;
  const int r    = lane & 15;
  const int head = blockIdx.y;
  const int qb   = blockIdx.x;

  const _Float16* qrow = qh + ((size_t)head * NTOK + qb * 16 + r) * HDIM;
  const _Float16* krow = kh + ((size_t)head * NTOK + qb * 16 + r) * HDIM;

  v8f c = {};
  c = wmma_f16(load_frag_row(qrow,      half), load_frag_row(krow,      half), c);
  c = wmma_f16(load_frag_row(qrow + 32, half), load_frag_row(krow + 32, half), c);

#pragma unroll
  for (int i = 0; i < 8; ++i) {
    int mrow = half * 8 + i;
    int mcol = r;
    float p = 0.f;
    if (mcol <= mrow) {                       // in-block causal
      size_t g = (size_t)head * NTOK + qb * 16 + mrow;
      p = __expf(c[i] - rowmax[g]) / rowsum[g];  // full-row normalization
    }
    P[mrow][mcol] = p;
  }
  for (int t = lane; t < 16 * HDIM; t += 32)
    V[t >> 6][t & 63] =
        vh[((size_t)head * NTOK + qb * 16 + (t >> 6)) * HDIM + (t & 63)];
  __syncthreads();

  for (int mrow = 0; mrow < 16; ++mrow) {
    for (int d = lane; d < HDIM; d += 32) {
      float acc = 0.f;
      for (int cc = 0; cc <= mrow; ++cc) acc += P[mrow][cc] * V[cc][d];
      wv[(size_t)(qb * 16 + mrow) * CDIM + head * HDIM + d] = acc;
    }
  }
}

// ---------------------------------------------------------------------------
// Elementwise kernels
// ---------------------------------------------------------------------------
__global__ void __launch_bounds__(256)
cvt_f32_f16(const float* __restrict__ in, _Float16* __restrict__ out, int n)
{
  int i = blockIdx.x * 256 + threadIdx.x;
  if (i < n) out[i] = (_Float16)in[i];
}

// q: scale by sqrt(HDIM)=8 and split to [H][N][D] f16; k -> f16; v -> f32.
__global__ void __launch_bounds__(256)
split_heads(const float* __restrict__ qf, const float* __restrict__ kvf,
            _Float16* __restrict__ qh, _Float16* __restrict__ kh,
            float* __restrict__ vh)
{
  size_t idx = (size_t)blockIdx.x * 256 + threadIdx.x;
  if (idx >= (size_t)NTOK * CDIM) return;
  int n = (int)(idx / CDIM), c = (int)(idx % CDIM);
  int h = c / HDIM, d = c % HDIM;
  size_t o = ((size_t)h * NTOK + n) * HDIM + d;
  qh[o] = (_Float16)(qf[idx] * 8.0f);                       // q * sqrt(D)
  kh[o] = (_Float16)kvf[(size_t)n * (2 * CDIM) + c];        // k
  vh[o] = kvf[(size_t)n * (2 * CDIM) + CDIM + c];           // v (keep f32)
}

// depthwise 1x1 conv == per-channel affine; emit f16 for the pointwise GEMM.
__global__ void __launch_bounds__(256)
dyn_affine(const float* __restrict__ wv, const float* __restrict__ dw_w,
           const float* __restrict__ dw_b, _Float16* __restrict__ dynh)
{
  size_t idx = (size_t)blockIdx.x * 256 + threadIdx.x;
  if (idx >= (size_t)NTOK * CDIM) return;
  int c = (int)(idx % CDIM);
  dynh[idx] = (_Float16)(wv[idx] * dw_w[c] + dw_b[c]);
}

// ---------------------------------------------------------------------------
// Host-side orchestration
// ---------------------------------------------------------------------------
extern "C" void kernel_launch(void* const* d_in, const int* in_sizes, int n_in,
                              void* d_out, int out_size, void* d_ws, size_t ws_size,
                              hipStream_t stream)
{
  (void)in_sizes; (void)n_in; (void)out_size; (void)ws_size;

  const float* x     = (const float*)d_in[0];   // [4096, 512]
  const float* q_w1  = (const float*)d_in[1];   // [32, 512]
  const float* q_w2  = (const float*)d_in[2];   // [512, 32]
  const float* kv_w1 = (const float*)d_in[3];   // [64, 512]
  const float* kv_w2 = (const float*)d_in[4];   // [1024, 64]
  const float* dw_w  = (const float*)d_in[5];   // [512]
  const float* dw_b  = (const float*)d_in[6];   // [512]
  const float* pw_w  = (const float*)d_in[7];   // [512, 512]
  const float* pw_b  = (const float*)d_in[8];   // [512]
  const float* p_w1  = (const float*)d_in[9];   // [32, 512]
  const float* p_w2  = (const float*)d_in[10];  // [512, 32]
  float*       out   = (float*)d_out;           // [4096, 512]

  // bump allocator over workspace
  char* ws = (char*)d_ws;
  auto alloc = [&](size_t bytes) -> void* {
    void* p = (void*)ws;
    ws += (bytes + 255) & ~(size_t)255;
    return p;
  };

  _Float16* xh      = (_Float16*)alloc((size_t)NTOK * CDIM * 2);
  _Float16* q_w1h   = (_Float16*)alloc((size_t)RANK * CDIM * 2);
  _Float16* q_w2h   = (_Float16*)alloc((size_t)CDIM * RANK * 2);
  _Float16* kv_w1h  = (_Float16*)alloc((size_t)2 * RANK * CDIM * 2);
  _Float16* kv_w2h  = (_Float16*)alloc((size_t)2 * CDIM * 2 * RANK * 2);
  _Float16* pw_wh   = (_Float16*)alloc((size_t)CDIM * CDIM * 2);
  _Float16* p_w1h   = (_Float16*)alloc((size_t)RANK * CDIM * 2);
  _Float16* p_w2h   = (_Float16*)alloc((size_t)CDIM * RANK * 2);

  float*    q_lr    = (float*)   alloc((size_t)NTOK * RANK * 4);
  _Float16* q_lrh   = (_Float16*)alloc((size_t)NTOK * RANK * 2);
  float*    kv_lr   = (float*)   alloc((size_t)NTOK * 2 * RANK * 4);
  _Float16* kv_lrh  = (_Float16*)alloc((size_t)NTOK * 2 * RANK * 2);
  float*    qf      = (float*)   alloc((size_t)NTOK * CDIM * 4);
  float*    kvf     = (float*)   alloc((size_t)NTOK * 2 * CDIM * 4);
  _Float16* qhB     = (_Float16*)alloc((size_t)NTOK * CDIM * 2);
  _Float16* khB     = (_Float16*)alloc((size_t)NTOK * CDIM * 2);
  float*    vhB     = (float*)   alloc((size_t)NTOK * CDIM * 4);
  float*    rowmax  = (float*)   alloc((size_t)NHEAD * NTOK * 4);
  float*    rowsum  = (float*)   alloc((size_t)NHEAD * NTOK * 4);
  float*    wv      = (float*)   alloc((size_t)NTOK * CDIM * 4);
  _Float16* dynh    = (_Float16*)alloc((size_t)NTOK * CDIM * 2);
  float*    y1      = (float*)   alloc((size_t)NTOK * CDIM * 4);
  _Float16* y1h     = (_Float16*)alloc((size_t)NTOK * CDIM * 2);
  float*    ylr     = (float*)   alloc((size_t)NTOK * RANK * 4);
  _Float16* ylrh    = (_Float16*)alloc((size_t)NTOK * RANK * 2);

  auto cvt = [&](const float* src, _Float16* dst, int n) {
    cvt_f32_f16<<<(n + 255) / 256, 256, 0, stream>>>(src, dst, n);
  };

  // f32 -> f16 staging
  cvt(x,     xh,     NTOK * CDIM);
  cvt(q_w1,  q_w1h,  RANK * CDIM);
  cvt(q_w2,  q_w2h,  CDIM * RANK);
  cvt(kv_w1, kv_w1h, 2 * RANK * CDIM);
  cvt(kv_w2, kv_w2h, 2 * CDIM * 2 * RANK);
  cvt(pw_w,  pw_wh,  CDIM * CDIM);
  cvt(p_w1,  p_w1h,  RANK * CDIM);
  cvt(p_w2,  p_w2h,  CDIM * RANK);

  dim3 wave(32);
  const int MT = NTOK / 16;  // 256 row tiles

  // q_lr = x @ q_w1^T                    [4096,32]   (NT=2 strip covers N=32)
  gemm_wmma_nt<2><<<dim3(MT, RANK / 32), wave, 0, stream>>>(
      xh, q_w1h, q_lr, q_lrh, nullptr, nullptr, NTOK, RANK, CDIM);
  // kv_lr = x @ kv_w1^T                  [4096,64]
  gemm_wmma_nt<4><<<dim3(MT, (2 * RANK) / 64), wave, 0, stream>>>(
      xh, kv_w1h, kv_lr, kv_lrh, nullptr, nullptr, NTOK, 2 * RANK, CDIM);
  // q = q_lr @ q_w2^T                    [4096,512]
  gemm_wmma_nt<4><<<dim3(MT, CDIM / 64), wave, 0, stream>>>(
      q_lrh, q_w2h, qf, nullptr, nullptr, nullptr, NTOK, CDIM, RANK);
  // kv = kv_lr @ kv_w2^T                 [4096,1024]
  gemm_wmma_nt<4><<<dim3(MT, (2 * CDIM) / 64), wave, 0, stream>>>(
      kv_lrh, kv_w2h, kvf, nullptr, nullptr, nullptr, NTOK, 2 * CDIM, 2 * RANK);

  // split heads (+ q * sqrt(D))
  split_heads<<<(NTOK * CDIM + 255) / 256, 256, 0, stream>>>(qf, kvf, qhB, khB, vhB);

  // flash-style full-row softmax stats, then diagonal-block P*V
  attn_stats<<<dim3(MT, NHEAD), wave, 0, stream>>>(qhB, khB, rowmax, rowsum);
  attn_diag_wv<<<dim3(MT, NHEAD), wave, 0, stream>>>(qhB, khB, vhB,
                                                     rowmax, rowsum, wv);

  // dynamic focus: per-channel affine, then pointwise conv + bias + residual
  dyn_affine<<<(NTOK * CDIM + 255) / 256, 256, 0, stream>>>(wv, dw_w, dw_b, dynh);
  gemm_wmma_nt<4><<<dim3(MT, CDIM / 64), wave, 0, stream>>>(
      dynh, pw_wh, y1, y1h, pw_b, x, NTOK, CDIM, CDIM);

  // output low-rank projection
  gemm_wmma_nt<2><<<dim3(MT, RANK / 32), wave, 0, stream>>>(
      y1h, p_w1h, ylr, ylrh, nullptr, nullptr, NTOK, RANK, CDIM);
  gemm_wmma_nt<4><<<dim3(MT, CDIM / 64), wave, 0, stream>>>(
      ylrh, p_w2h, out, nullptr, nullptr, nullptr, NTOK, CDIM, RANK);
}
